// MultiHeadAttentionBlock_28535762714875
// MI455X (gfx1250) — matrix operand order repaired and checked by reading, not compiled, not run on hardware
//
#include <hip/hip_runtime.h>
#include <hip/hip_bf16.h>

// ---------------------------------------------------------------------------
// MHA block for MI455X (gfx1250, wave32, WMMA, async-to-LDS).
// Phase 0: stream-convert X (3x) and W (3x) fp32 -> bf16 in workspace.
// Phase 1: q/k/v projection GEMMs (bf16 WMMA, async W staging), bf16 out.
// Phase 2: fused flash attention with online softmax (bf16 WMMA).
//   Reference role swap: queries = v-proj, keys = k-proj, values = q-proj.
//   scale = 1/sqrt(EMB_DIM) = 1/32.
// ---------------------------------------------------------------------------

#define HEADS    16
#define HEAD_DIM 64
#define EMB      1024
#define SEQ      2048
#define BATCH    2
#define MTOT     (BATCH * SEQ)   // 4096 tokens

typedef __attribute__((ext_vector_type(16))) __bf16          v16bf;
typedef __attribute__((ext_vector_type(16))) unsigned short  v16u;
typedef __attribute__((ext_vector_type(8)))  float           v8f;
typedef __attribute__((ext_vector_type(8)))  unsigned short  u16x8;
typedef __attribute__((ext_vector_type(4)))  unsigned int    u32x4;
typedef __attribute__((ext_vector_type(4)))  float           f32x4;

union FragU {
    v16u           v;
    u32x4          q[2];
    unsigned short s[16];
};

// hardware f32 -> bf16 (backend emits native cvt on gfx1250)
__device__ __forceinline__ unsigned short f2bf(float f) {
    return __builtin_bit_cast(unsigned short, (__bf16)f);
}

__device__ __forceinline__ v8f vzero8() {
    v8f z;
#pragma unroll
    for (int i = 0; i < 8; ++i) z[i] = 0.0f;
    return z;
}

// reductions across a 16-lane half of the wave (C-layout row reductions)
__device__ __forceinline__ float rmax16(float v) {
    v = fmaxf(v, __shfl_xor(v, 1));
    v = fmaxf(v, __shfl_xor(v, 2));
    v = fmaxf(v, __shfl_xor(v, 4));
    v = fmaxf(v, __shfl_xor(v, 8));
    return v;
}
__device__ __forceinline__ float rsum16(float v) {
    v += __shfl_xor(v, 1);
    v += __shfl_xor(v, 2);
    v += __shfl_xor(v, 4);
    v += __shfl_xor(v, 8);
    return v;
}

#define BF16_WMMA(A, B, C)                                                     \
    __builtin_amdgcn_wmma_f32_16x16x32_bf16(                                   \
        false, __builtin_bit_cast(v16bf, (A)),                                 \
        false, __builtin_bit_cast(v16bf, (B)),                                 \
        (short)0, (C), false, false)

// ---- CDNA5 async global->LDS path (ASYNCcnt), guarded ---------------------
#if defined(__has_builtin)
#if __has_builtin(__builtin_amdgcn_global_load_async_to_lds_b128) && \
    __has_builtin(__builtin_amdgcn_s_wait_asynccnt)
#define HAVE_ASYNC_LDS 1
#endif
#endif
#ifndef HAVE_ASYNC_LDS
#define HAVE_ASYNC_LDS 0
#endif

#define AS1 __attribute__((address_space(1)))
#define AS3 __attribute__((address_space(3)))

// GCC-style v4i matching the builtin's parameter type exactly
typedef int gv4i __attribute__((vector_size(16)));

// Value-correct AS conversions via integer cast:
// generic-LDS addr[31:0] == LDS offset; generic-global == physical address.
__device__ __forceinline__ AS3 gv4i* lds_cast(void* p) {
    return (AS3 gv4i*)(unsigned long long)(p);
}
__device__ __forceinline__ AS1 gv4i* gbl_cast(const void* p) {
    return (AS1 gv4i*)(unsigned long long)(p);
}

// copy 32 bytes global(bf16) -> LDS, async when available
__device__ __forceinline__ void stage32B(const unsigned short* g, unsigned short* l) {
#if HAVE_ASYNC_LDS
    __builtin_amdgcn_global_load_async_to_lds_b128(gbl_cast(g), lds_cast(l), 0, 0);
    __builtin_amdgcn_global_load_async_to_lds_b128(gbl_cast(g), lds_cast(l), 16, 0);
#else
    *(u32x4*)(l)     = *(const u32x4*)(g);
    *(u32x4*)(l + 8) = *(const u32x4*)(g + 8);
#endif
}
__device__ __forceinline__ void stage_wait() {
#if HAVE_ASYNC_LDS
    __builtin_amdgcn_s_wait_asynccnt(0);
#endif
}

// ---------------------------------------------------------------------------
// Phase 0: streaming fp32 -> bf16 conversion (n must be a multiple of 8)
// ---------------------------------------------------------------------------
__launch_bounds__(256)
__global__ void cvt_f32_bf16(const float* __restrict__ in,
                             unsigned short* __restrict__ out, int n) {
    const int i = (blockIdx.x * 256 + threadIdx.x) * 8;
    if (i >= n) return;
    f32x4 a = *(const f32x4*)(in + i);
    f32x4 b = *(const f32x4*)(in + i + 4);
    u16x8 r;
    r[0] = f2bf(a.x); r[1] = f2bf(a.y); r[2] = f2bf(a.z); r[3] = f2bf(a.w);
    r[4] = f2bf(b.x); r[5] = f2bf(b.y); r[6] = f2bf(b.z); r[7] = f2bf(b.w);
    *(u16x8*)(out + i) = r;
}

// ---------------------------------------------------------------------------
// Phase 1: OUT[4096,1024](bf16) = Xb[4096,1024] @ Wb^T[1024,1024] + bias
// 128 threads = 4 waves; WG tile 64(M) x 64(N); wave tile 16(M) x 64(N).
// ---------------------------------------------------------------------------
#define SW_STRIDE 40   // ushorts per staged W row (32 data + pad; 80B rows)

__launch_bounds__(128)
__global__ void proj_gemm_bf16(const unsigned short* __restrict__ Xb,
                               const unsigned short* __restrict__ Wb,
                               const float* __restrict__ bias,
                               unsigned short* __restrict__ OUT) {
    __shared__ unsigned short sW[64 * SW_STRIDE];

    const int tid  = threadIdx.x;
    const int wave = tid >> 5, lane = tid & 31;
    const int half = lane >> 4, ln = lane & 15;
    const int mBlock = blockIdx.x * 64;
    const int nBlock = blockIdx.y * 64;
    const int mRow   = mBlock + wave * 16 + ln;   // A-fragment row of this lane

    v8f acc[4];
#pragma unroll
    for (int j = 0; j < 4; ++j) acc[j] = vzero8();

    const int nStage = tid >> 1;           // 0..63
    const int kStage = (tid & 1) * 16;     // 0 or 16
    const unsigned short* wsrc = Wb + (size_t)(nBlock + nStage) * EMB + kStage;
    unsigned short*       wdst = &sW[nStage * SW_STRIDE + kStage];
    const unsigned short* xrow = Xb + (size_t)mRow * EMB;

    for (int k0 = 0; k0 < EMB; k0 += 32) {
        __syncthreads();                    // previous tile fully consumed
        stage32B(wsrc + k0, wdst);          // 64x32 W tile, async -> LDS
        if (k0 + 32 < EMB) __builtin_prefetch(wsrc + k0 + 32, 0, 1);
        stage_wait();
        __syncthreads();

        // A fragment (16x32 bf16 ISA layout): e<8 -> K=half*8+e,
        // e>=8 -> K=16+half*8+(e-8); straight b128 loads from bf16 X
        FragU a;
        a.q[0] = *(const u32x4*)(xrow + k0 + half * 8);
        a.q[1] = *(const u32x4*)(xrow + k0 + 16 + half * 8);

        // load all 4 B fragments, then issue 4 back-to-back WMMAs
        FragU bf[4];
#pragma unroll
        for (int j = 0; j < 4; ++j) {
            const unsigned short* bs = &sW[(j * 16 + ln) * SW_STRIDE + half * 8];
            bf[j].q[0] = *(const u32x4*)(bs);
            bf[j].q[1] = *(const u32x4*)(bs + 16);
        }
#pragma unroll
        for (int j = 0; j < 4; ++j)
            acc[j] = BF16_WMMA(a.v, bf[j].v, acc[j]);
    }

    // C layout: element (m = r + 8*half, n = j*16 + ln); add bias, store bf16
#pragma unroll
    for (int j = 0; j < 4; ++j) {
        const int n    = nBlock + j * 16 + ln;
        const float bv = bias[n];
#pragma unroll
        for (int r = 0; r < 8; ++r) {
            const int m = mBlock + wave * 16 + r + 8 * half;
            OUT[(size_t)m * EMB + n] = f2bf(acc[j][r] + bv);
        }
    }
}

// ---------------------------------------------------------------------------
// Phase 2: flash attention. One WG (4 waves) per (b, h, 64 q rows); each wave
// owns 16 q rows. Keys in chunks of 32 with online softmax.
// ---------------------------------------------------------------------------
#define SK_STRIDE 72   // K tile row stride (64 data + pad; 144B rows)
#define SV_STRIDE 40   // V^T tile row stride (32 data + pad; 80B rows)
#define SP_STRIDE 40   // P tile row stride (32 data + pad; 80B rows)

__launch_bounds__(128)
__global__ void flash_attn(const unsigned short* __restrict__ Qp,   // queries (= v-proj)
                           const unsigned short* __restrict__ Kp,   // keys    (= k-proj)
                           const unsigned short* __restrict__ Vp,   // values  (= q-proj)
                           float* __restrict__ OUT) {
    __shared__ unsigned short sK [32 * SK_STRIDE];     // [key][d]
    __shared__ unsigned short sVt[64 * SV_STRIDE];     // [d][key] (transposed)
    __shared__ unsigned short sP [4 * 16 * SP_STRIDE]; // per-wave P (16 x 32)

    const int tid  = threadIdx.x;
    const int wave = tid >> 5, lane = tid & 31;
    const int half = lane >> 4, ln = lane & 15;
    const int b = blockIdx.z, h = blockIdx.y;
    const int q0 = blockIdx.x * 64 + wave * 16;
    const float scale = 0.03125f;                      // 1/sqrt(EMB_DIM)

    // Preload Q A-fragments for K-dim tiles d=[0,32) and d=[32,64)
    FragU aQ[2];
    {
        const unsigned short* qrow =
            Qp + (size_t)(b * SEQ + q0 + ln) * EMB + h * HEAD_DIM;
#pragma unroll
        for (int t = 0; t < 2; ++t) {
            aQ[t].q[0] = *(const u32x4*)(qrow + t * 32 + half * 8);
            aQ[t].q[1] = *(const u32x4*)(qrow + t * 32 + 16 + half * 8);
        }
    }

    v8f o[4];
#pragma unroll
    for (int j = 0; j < 4; ++j) o[j] = vzero8();
    float mrow[8], lrow[8];
#pragma unroll
    for (int r = 0; r < 8; ++r) { mrow[r] = -3.0e38f; lrow[r] = 0.0f; }

    // staging assignment: thread -> (key 0..31, 16-wide d group)
    const int sKey = tid >> 2;
    const int sD   = (tid & 3) * 16;
    const unsigned short* kbase =
        Kp + (size_t)b * SEQ * EMB + (size_t)sKey * EMB + h * HEAD_DIM + sD;
    const unsigned short* vbase =
        Vp + (size_t)b * SEQ * EMB + (size_t)sKey * EMB + h * HEAD_DIM + sD;
    unsigned short* sPw = &sP[wave * 16 * SP_STRIDE];

    for (int key0 = 0; key0 < SEQ; key0 += 32) {
        __syncthreads();   // previous chunk fully consumed before overwrite
        {   // K chunk [32 x 64]: async straight copy into LDS
            const unsigned short* ks = kbase + (size_t)key0 * EMB;
            stage32B(ks, &sK[sKey * SK_STRIDE + sD]);

            // V chunk [64 x 32] staged transposed (per-lane scatter)
            const unsigned short* vs = vbase + (size_t)key0 * EMB;
            u32x4 va = *(const u32x4*)(vs);
            u32x4 vb = *(const u32x4*)(vs + 8);
            const unsigned short* va_s = (const unsigned short*)&va;
            const unsigned short* vb_s = (const unsigned short*)&vb;
#pragma unroll
            for (int i = 0; i < 8; ++i) {
                sVt[(sD + i)     * SV_STRIDE + sKey] = va_s[i];
                sVt[(sD + 8 + i) * SV_STRIDE + sKey] = vb_s[i];
            }
            if (key0 + 32 < SEQ) {
                __builtin_prefetch(ks + (size_t)32 * EMB, 0, 1);
                __builtin_prefetch(vs + (size_t)32 * EMB, 0, 1);
            }
        }
        stage_wait();
        __syncthreads();

        // ---- S = Q . K^T for 32 keys: load all B frags, then 4 WMMAs ----
        FragU bk[2][2];   // [key group][d tile]
#pragma unroll
        for (int g = 0; g < 2; ++g)
#pragma unroll
            for (int t = 0; t < 2; ++t) {
                const unsigned short* p =
                    &sK[(g * 16 + ln) * SK_STRIDE + t * 32 + half * 8];
                bk[g][t].q[0] = *(const u32x4*)(p);
                bk[g][t].q[1] = *(const u32x4*)(p + 16);
            }
        v8f s0 = vzero8(), s1 = vzero8();
        s0 = BF16_WMMA(aQ[0].v, bk[0][0].v, s0);
        s1 = BF16_WMMA(aQ[0].v, bk[1][0].v, s1);
        s0 = BF16_WMMA(aQ[1].v, bk[0][1].v, s0);
        s1 = BF16_WMMA(aQ[1].v, bk[1][1].v, s1);

        // ---- online softmax over these 32 keys (row m = r + 8*half) ----
#pragma unroll
        for (int r = 0; r < 8; ++r) {
            float a0 = s0[r] * scale;
            float a1 = s1[r] * scale;
            float mx    = rmax16(fmaxf(a0, a1));
            float mnew  = fmaxf(mrow[r], mx);
            float alpha = __expf(mrow[r] - mnew);
            float p0 = __expf(a0 - mnew);
            float p1 = __expf(a1 - mnew);
            lrow[r] = lrow[r] * alpha + rsum16(p0 + p1);
            mrow[r] = mnew;
#pragma unroll
            for (int j = 0; j < 4; ++j) o[j][r] *= alpha;
            const int m = r + 8 * half;
            sPw[m * SP_STRIDE + ln]      = f2bf(p0);   // keys [key0, key0+16)
            sPw[m * SP_STRIDE + 16 + ln] = f2bf(p1);   // keys [key0+16, key0+32)
        }
        // wave-private C->A layout shuffle through LDS: commit stores
        asm volatile("s_wait_dscnt 0" ::: "memory");

        // ---- O += P @ V ----
        FragU aP;
        {
            const unsigned short* pr = &sPw[ln * SP_STRIDE + half * 8];
            aP.q[0] = *(const u32x4*)(pr);        // K = half*8 + e
            aP.q[1] = *(const u32x4*)(pr + 16);   // K = 16 + half*8 + e
        }
        FragU bV[4];
#pragma unroll
        for (int j = 0; j < 4; ++j) {
            const unsigned short* vp2 = &sVt[(j * 16 + ln) * SV_STRIDE + half * 8];
            bV[j].q[0] = *(const u32x4*)(vp2);
            bV[j].q[1] = *(const u32x4*)(vp2 + 16);
        }
#pragma unroll
        for (int j = 0; j < 4; ++j)
            o[j] = BF16_WMMA(aP.v, bV[j].v, o[j]);
    }

    // ---- epilogue: O / l, fp32 out[b, q, h*64 + d] ----
    float* obase = OUT + (size_t)(b * SEQ + q0) * EMB + h * HEAD_DIM;
#pragma unroll
    for (int r = 0; r < 8; ++r) {
        const float inv = 1.0f / lrow[r];
        const int m = r + 8 * half;
#pragma unroll
        for (int j = 0; j < 4; ++j)
            obase[(size_t)m * EMB + j * 16 + ln] = o[j][r] * inv;
    }
}

// ---------------------------------------------------------------------------
extern "C" void kernel_launch(void* const* d_in, const int* in_sizes, int n_in,
                              void* d_out, int out_size, void* d_ws, size_t ws_size,
                              hipStream_t stream) {
    (void)in_sizes; (void)n_in; (void)out_size; (void)ws_size;

    const float* Xq = (const float*)d_in[0];
    const float* Xk = (const float*)d_in[1];
    const float* Xv = (const float*)d_in[2];
    const float* Wq = (const float*)d_in[3];
    const float* bq = (const float*)d_in[4];
    const float* Wk = (const float*)d_in[5];
    const float* bk = (const float*)d_in[6];
    const float* Wv = (const float*)d_in[7];
    const float* bv = (const float*)d_in[8];

    const size_t NX = (size_t)MTOT * EMB;   // 4096*1024
    const size_t NW = (size_t)EMB * EMB;    // 1024*1024

    // workspace layout (bf16/ushort): projections, converted X, converted W
    unsigned short* qp  = (unsigned short*)d_ws;
    unsigned short* kp  = qp  + NX;
    unsigned short* vp  = kp  + NX;
    unsigned short* xqb = vp  + NX;
    unsigned short* xkb = xqb + NX;
    unsigned short* xvb = xkb + NX;
    unsigned short* wqb = xvb + NX;
    unsigned short* wkb = wqb + NW;
    unsigned short* wvb = wkb + NW;

    // Phase 0: stream conversions fp32 -> bf16
    const int gx = (int)(NX / 8 / 256);     // 2048 blocks
    const int gw = (int)(NW / 8 / 256);     // 512 blocks
    cvt_f32_bf16<<<gx, 256, 0, stream>>>(Xq, xqb, (int)NX);
    cvt_f32_bf16<<<gx, 256, 0, stream>>>(Xk, xkb, (int)NX);
    cvt_f32_bf16<<<gx, 256, 0, stream>>>(Xv, xvb, (int)NX);
    cvt_f32_bf16<<<gw, 256, 0, stream>>>(Wq, wqb, (int)NW);
    cvt_f32_bf16<<<gw, 256, 0, stream>>>(Wk, wkb, (int)NW);
    cvt_f32_bf16<<<gw, 256, 0, stream>>>(Wv, wvb, (int)NW);

    // Phase 1: projections
    dim3 gProj(MTOT / 64, EMB / 64, 1);     // (64, 16)
    proj_gemm_bf16<<<gProj, 128, 0, stream>>>(xqb, wqb, bq, qp);
    proj_gemm_bf16<<<gProj, 128, 0, stream>>>(xkb, wkb, bk, kp);
    proj_gemm_bf16<<<gProj, 128, 0, stream>>>(xvb, wvb, bv, vp);

    // Phase 2: attention (role swap: queries = v-proj, values = q-proj)
    dim3 gAttn(SEQ / 64, HEADS, BATCH);     // (32, 16, 2)
    flash_attn<<<gAttn, 128, 0, stream>>>(vp, kp, qp, (float*)d_out);
}
